// LSTM_Semenuita_21474836480119
// MI455X (gfx1250) — matrix operand order repaired and checked by reading, not compiled, not run on hardware
//
#include <hip/hip_runtime.h>

// Problem constants
constexpr int kB  = 64;
constexpr int kT  = 512;
constexpr int kD  = 512;
constexpr int kH  = 1024;
constexpr int k4H = 4096;

typedef __attribute__((ext_vector_type(16))) __bf16        v16bf;
typedef __attribute__((ext_vector_type(8)))  float         v8f;
typedef __attribute__((ext_vector_type(4)))  unsigned int  u32x4;

union BFVec { v16bf v; u32x4 q[2]; };

__device__ __forceinline__ unsigned short f2bf(float f) {
    unsigned int u = __builtin_bit_cast(unsigned int, f);
    u += 0x7FFFu + ((u >> 16) & 1u);   // round-to-nearest-even
    return (unsigned short)(u >> 16);
}

__device__ __forceinline__ float sigmoid_fast(float x) {
    x = fminf(fmaxf(x, -30.0f), 30.0f);
    return 1.0f / (1.0f + __expf(-x));
}
__device__ __forceinline__ float tanh_fast(float x) {
    x = fminf(fmaxf(x, -15.0f), 15.0f);
    float e = __expf(2.0f * x);
    return (e - 1.0f) / (e + 1.0f);
}

// ---------------- fp32 -> bf16 conversion (grid-stride) ----------------
__global__ void k_cvt_bf16(const float* __restrict__ src,
                           unsigned short* __restrict__ dst, int n) {
    int i = blockIdx.x * blockDim.x + threadIdx.x;
    int stride = gridDim.x * blockDim.x;
    for (; i < n; i += stride) dst[i] = f2bf(src[i]);
}

// ---------------- zero h0 (bf16) and c0 (f32, tiled) ----------------
__global__ void k_init_state(unsigned short* __restrict__ hbf,
                             float* __restrict__ c, int n) {
    int i = blockIdx.x * blockDim.x + threadIdx.x;
    if (i < n) { hbf[i] = 0; c[i] = 0.0f; }
}

// ---------------- Phase 1: xg = x @ Wx^T + bx ----------------
// M ordering: m = t*64 + b  (16-row tile == one timestep, 16 batches).
// 4x2 register blocking: one wave computes a 64x32 output macro-tile
// (one full timestep x 2 column tiles): 8 accumulators (64 VGPRs),
// 8 WMMAs per K-chunk from 12 b128 loads. #pragma unroll 1 keeps the
// K-loop rolled so accumulators stay register-carried (no spills).
// Output stored in native WMMA C-fragment layout:
//   xg[((t*256 + ct)*4 + bt)*256 + lane*8 + i]   (ct = gate*64 + ntile)
__global__ void k_xg_gemm(const unsigned short* __restrict__ xb,   // [B*T, D] bf16, row = b*T+t
                          const unsigned short* __restrict__ wxb,  // [4H, D]  bf16
                          const float* __restrict__ bx,            // [4H]
                          float* __restrict__ xg)                  // tiled, T*B*4H f32
{
    const int wave  = threadIdx.x >> 5;
    const int lane  = threadIdx.x & 31;
    const int s     = blockIdx.x * 4 + wave;   // 65536 waves
    const int t     = s >> 7;                  // 512 macro-m (one timestep each)
    const int ct0   = (s & 127) << 1;          // macro-n: 2 column tiles
    const int lhalf = lane >> 4;
    const int lmod  = lane & 15;
    const int aoff  = lhalf << 3;   // 0 / 8  bf16
    const int boff  = lhalf << 3;   // 0 / 8  bf16

    const unsigned short* arow0 = xb + ((size_t)( 0 + lmod) * kT + t) * kD;
    const unsigned short* arow1 = xb + ((size_t)(16 + lmod) * kT + t) * kD;
    const unsigned short* arow2 = xb + ((size_t)(32 + lmod) * kT + t) * kD;
    const unsigned short* arow3 = xb + ((size_t)(48 + lmod) * kT + t) * kD;
    const unsigned short* bcol0 = wxb + (size_t)((ct0 + 0) * 16 + lmod) * kD;
    const unsigned short* bcol1 = wxb + (size_t)((ct0 + 1) * 16 + lmod) * kD;

    v8f acc00 = {}, acc01 = {};
    v8f acc10 = {}, acc11 = {};
    v8f acc20 = {}, acc21 = {};
    v8f acc30 = {}, acc31 = {};

    #pragma unroll 1
    for (int k0 = 0; k0 < kD; k0 += 32) {
        BFVec a0, a1, a2, a3, w0, w1;
        a0.q[0] = *(const u32x4*)(arow0 + k0 + aoff);
        a0.q[1] = *(const u32x4*)(arow0 + k0 + 16 + aoff);
        a1.q[0] = *(const u32x4*)(arow1 + k0 + aoff);
        a1.q[1] = *(const u32x4*)(arow1 + k0 + 16 + aoff);
        a2.q[0] = *(const u32x4*)(arow2 + k0 + aoff);
        a2.q[1] = *(const u32x4*)(arow2 + k0 + 16 + aoff);
        a3.q[0] = *(const u32x4*)(arow3 + k0 + aoff);
        a3.q[1] = *(const u32x4*)(arow3 + k0 + 16 + aoff);
        w0.q[0] = *(const u32x4*)(bcol0 + k0 + boff);
        w0.q[1] = *(const u32x4*)(bcol0 + k0 + 16 + boff);
        w1.q[0] = *(const u32x4*)(bcol1 + k0 + boff);
        w1.q[1] = *(const u32x4*)(bcol1 + k0 + 16 + boff);
        acc00 = __builtin_amdgcn_wmma_f32_16x16x32_bf16(
                    false, a0.v, false, w0.v, (short)0, acc00, false, false);
        acc01 = __builtin_amdgcn_wmma_f32_16x16x32_bf16(
                    false, a0.v, false, w1.v, (short)0, acc01, false, false);
        acc10 = __builtin_amdgcn_wmma_f32_16x16x32_bf16(
                    false, a1.v, false, w0.v, (short)0, acc10, false, false);
        acc11 = __builtin_amdgcn_wmma_f32_16x16x32_bf16(
                    false, a1.v, false, w1.v, (short)0, acc11, false, false);
        acc20 = __builtin_amdgcn_wmma_f32_16x16x32_bf16(
                    false, a2.v, false, w0.v, (short)0, acc20, false, false);
        acc21 = __builtin_amdgcn_wmma_f32_16x16x32_bf16(
                    false, a2.v, false, w1.v, (short)0, acc21, false, false);
        acc30 = __builtin_amdgcn_wmma_f32_16x16x32_bf16(
                    false, a3.v, false, w0.v, (short)0, acc30, false, false);
        acc31 = __builtin_amdgcn_wmma_f32_16x16x32_bf16(
                    false, a3.v, false, w1.v, (short)0, acc31, false, false);
    }

    const float bias0 = bx[(ct0 + 0) * 16 + lmod];
    const float bias1 = bx[(ct0 + 1) * 16 + lmod];
    v8f r[4][2] = {{acc00, acc01}, {acc10, acc11}, {acc20, acc21}, {acc30, acc31}};
    #pragma unroll
    for (int bi = 0; bi < 4; ++bi) {
        #pragma unroll
        for (int ci = 0; ci < 2; ++ci) {
            v8f v = r[bi][ci];
            const float bias = ci ? bias1 : bias0;
            #pragma unroll
            for (int i = 0; i < 8; ++i) v[i] += bias;
            float* dst = xg + (((size_t)(t * 256 + ct0 + ci) * 4 + bi) << 8) + lane * 8;
            *(v8f*)dst = v;   // one contiguous 32B store per lane
        }
    }
}

// ---------------- Phase 2: one recurrent step ----------------
// grid = 64 blocks (h-tiles), 4 waves/block (batch-tiles). Each wave:
// 16x16 h-tile, 4 gate accumulators seeded from tiled xg (contiguous
// 32B loads), K=1024 -> 128 WMMAs with shared A (h_{t-1}) fragment.
__global__ void k_lstm_step(const float* __restrict__ xg_t,          // tiled [256][4][32][8] f32
                            const unsigned short* __restrict__ whb,  // [4H, H] bf16
                            const unsigned short* __restrict__ h_in, // [B, H]  bf16
                            unsigned short* __restrict__ h_out,      // [B, H]  bf16
                            float* __restrict__ c,                   // tiled [64][4][32][8] f32
                            float* __restrict__ out,                 // [B, T, H] f32
                            int t)
{
    const int wave   = threadIdx.x >> 5;
    const int lane   = threadIdx.x & 31;
    const int n_tile = blockIdx.x;       // 0..63
    const int b_tile = wave;             // 0..3
    const int h0     = n_tile << 4;
    const int b0     = b_tile << 4;
    const int lhalf  = lane >> 4;
    const int lmod   = lane & 15;

    // Seed accumulators from tiled xg: contiguous 32B per lane per gate.
    v8f acc0 = *(const v8f*)(xg_t + (((size_t)(0 * 64 + n_tile) * 4 + b_tile) << 8) + lane * 8);
    v8f acc1 = *(const v8f*)(xg_t + (((size_t)(1 * 64 + n_tile) * 4 + b_tile) << 8) + lane * 8);
    v8f acc2 = *(const v8f*)(xg_t + (((size_t)(2 * 64 + n_tile) * 4 + b_tile) << 8) + lane * 8);
    v8f acc3 = *(const v8f*)(xg_t + (((size_t)(3 * 64 + n_tile) * 4 + b_tile) << 8) + lane * 8);

    const unsigned short* arow  = h_in + (size_t)(b0 + lmod) * kH;
    const unsigned short* bcol0 = whb + (size_t)(0 * kH + h0 + lmod) * kH;
    const unsigned short* bcol1 = whb + (size_t)(1 * kH + h0 + lmod) * kH;
    const unsigned short* bcol2 = whb + (size_t)(2 * kH + h0 + lmod) * kH;
    const unsigned short* bcol3 = whb + (size_t)(3 * kH + h0 + lmod) * kH;
    const int aoff = lhalf << 4;
    const int boff = lhalf << 4;

    #pragma unroll 2
    for (int k0 = 0; k0 < kH; k0 += 32) {
        BFVec a, w0, w1, w2, w3;
        a.q[0]  = *(const u32x4*)(arow + k0 + aoff);
        a.q[1]  = *(const u32x4*)(arow + k0 + aoff + 8);
        w0.q[0] = *(const u32x4*)(bcol0 + k0 + boff);
        w0.q[1] = *(const u32x4*)(bcol0 + k0 + boff + 8);
        w1.q[0] = *(const u32x4*)(bcol1 + k0 + boff);
        w1.q[1] = *(const u32x4*)(bcol1 + k0 + boff + 8);
        w2.q[0] = *(const u32x4*)(bcol2 + k0 + boff);
        w2.q[1] = *(const u32x4*)(bcol2 + k0 + boff + 8);
        w3.q[0] = *(const u32x4*)(bcol3 + k0 + boff);
        w3.q[1] = *(const u32x4*)(bcol3 + k0 + boff + 8);
        if (k0 + 32 < kH) __builtin_prefetch(arow + k0 + 32 + aoff, 0, 0);
        acc0 = __builtin_amdgcn_wmma_f32_16x16x32_bf16(
                   false, a.v, false, w0.v, (short)0, acc0, false, false);
        acc1 = __builtin_amdgcn_wmma_f32_16x16x32_bf16(
                   false, a.v, false, w1.v, (short)0, acc1, false, false);
        acc2 = __builtin_amdgcn_wmma_f32_16x16x32_bf16(
                   false, a.v, false, w2.v, (short)0, acc2, false, false);
        acc3 = __builtin_amdgcn_wmma_f32_16x16x32_bf16(
                   false, a.v, false, w3.v, (short)0, acc3, false, false);
    }

    // Gate nonlinearities; cell state kept in the same tiled fragment layout.
    float* cp = c + (((size_t)(n_tile * 4 + b_tile)) << 8) + lane * 8;
    #pragma unroll
    for (int i = 0; i < 8; ++i) {
        int m = b0 + i + (lhalf << 3);     // batch index
        int n = h0 + lmod;                 // hidden index
        float iv = sigmoid_fast(acc0[i]);
        float fv = sigmoid_fast(acc1[i]);
        float gv = tanh_fast   (acc2[i]);
        float ov = sigmoid_fast(acc3[i]);
        float cn = fv * cp[i] + iv * gv;
        cp[i] = cn;
        float hn = ov * tanh_fast(cn);
        out[(size_t)(m * kT + t) * kH + n] = hn;   // lanes 0-15 contiguous in h
        h_out[(size_t)m * kH + n] = f2bf(hn);
    }
}

// ---------------- tail: (hT, cT) appended to d_out ----------------
__global__ void k_tail(const float* __restrict__ c, float* __restrict__ d_out) {
    int idx = blockIdx.x * blockDim.x + threadIdx.x;
    if (idx >= kB * kH) return;
    const size_t BTH = (size_t)kB * kT * kH;
    // hT: copy from output at t = T-1
    {
        int b = idx >> 10, h = idx & (kH - 1);
        d_out[BTH + idx] = d_out[(size_t)(b * kT + (kT - 1)) * kH + h];
    }
    // cT: unscramble tiled cell-state fragment layout
    {
        int e    = idx & 7;
        int lane = (idx >> 3) & 31;
        int bt   = (idx >> 8) & 3;
        int nt   = idx >> 10;
        int b    = bt * 16 + e + ((lane >> 4) << 3);
        int h    = nt * 16 + (lane & 15);
        d_out[BTH + kB * kH + (size_t)b * kH + h] = c[idx];
    }
}

extern "C" void kernel_launch(void* const* d_in, const int* in_sizes, int n_in,
                              void* d_out, int out_size, void* d_ws, size_t ws_size,
                              hipStream_t stream) {
    const float* x  = (const float*)d_in[0];   // [B, T, D]
    const float* Wx = (const float*)d_in[1];   // [4H, D]
    const float* bx = (const float*)d_in[2];   // [4H]
    const float* Wh = (const float*)d_in[3];   // [4H, H]
    float* out = (float*)d_out;

    // Workspace carve-up (256B aligned slices)
    char* ws = (char*)d_ws;
    size_t off = 0;
    auto carve = [&](size_t bytes) -> void* {
        off = (off + 255) & ~(size_t)255;
        void* p = ws + off;
        off += bytes;
        return p;
    };
    float*          xg  = (float*)         carve((size_t)kT * kB * k4H * 4); // 536 MB, tiled
    unsigned short* xb  = (unsigned short*)carve((size_t)kB * kT * kD * 2);  // 33.5 MB
    unsigned short* wxb = (unsigned short*)carve((size_t)k4H * kD * 2);      // 4.2 MB
    unsigned short* whb = (unsigned short*)carve((size_t)k4H * kH * 2);      // 8.4 MB
    unsigned short* hA  = (unsigned short*)carve((size_t)kB * kH * 2);
    unsigned short* hB  = (unsigned short*)carve((size_t)kB * kH * 2);
    float*          cst = (float*)         carve((size_t)kB * kH * 4);       // tiled
    (void)ws_size; (void)in_sizes; (void)n_in; (void)out_size;

    // bf16 conversions
    k_cvt_bf16<<<4096, 256, 0, stream>>>(x,  xb,  kB * kT * kD);
    k_cvt_bf16<<<2048, 256, 0, stream>>>(Wx, wxb, k4H * kD);
    k_cvt_bf16<<<2048, 256, 0, stream>>>(Wh, whb, k4H * kH);
    k_init_state<<<(kB * kH + 255) / 256, 256, 0, stream>>>(hA, cst, kB * kH);

    // Phase 1: 512 t-macros x 128 n-macros = 65536 waves, 4 waves/block.
    k_xg_gemm<<<16384, 128, 0, stream>>>(xb, wxb, bx, xg);

    // Phase 2: 512 dependent steps; ping-pong h buffers.
    for (int t = 0; t < kT; ++t) {
        const unsigned short* hin = (t & 1) ? hB : hA;
        unsigned short*      hout = (t & 1) ? hA : hB;
        k_lstm_step<<<64, 128, 0, stream>>>(xg + (size_t)t * kB * k4H,
                                            whb, hin, hout, cst, out, t);
    }

    // Tail: (hT, cT)
    k_tail<<<(kB * kH + 255) / 256, 256, 0, stream>>>(cst, out);
}
